// PointCNP_17952963297844
// MI455X (gfx1250) — matrix operand order repaired and checked by reading, not compiled, not run on hardware
//
#include <hip/hip_runtime.h>
#include <hip/hip_bf16.h>
#include <cstdint>
#include <cstddef>

typedef __attribute__((ext_vector_type(16))) _Float16 v16h;
typedef __attribute__((ext_vector_type(8)))  float    v8f;

#define B_    64
#define NCTX  400
#define NTGT  784
#define M_    784      // GRID*GRID grid points
#define NBHD_ 9
#define GRID_ 28

// linspace(-14, 14, 28)[i]
__device__ __forceinline__ float lin28(int i) {
    return -14.0f + (28.0f / 27.0f) * (float)i;
}
__device__ __forceinline__ float softplusf(float x) {
    return (x > 20.0f) ? x : log1pf(__expf(x));
}

// ---------------------------------------------------------------------------
// Kernel 1: 9-nearest-neighbor indices on the 28x28 grid (batch-invariant).
// Replicates stable top_k(-d2, 9): ascending (d2, index) selection.
// ---------------------------------------------------------------------------
__global__ void k_geometry(int* __restrict__ idx) {
    int m = blockIdx.x * blockDim.x + threadIdx.x;
    if (m >= M_) return;
    float gx = lin28(m / GRID_), gy = lin28(m % GRID_);
    float pd = -1.0f; int pn = -1;
    for (int s = 0; s < NBHD_; ++s) {
        float bd = 3.4e38f; int bn = 0x7fffffff;
        int n = 0;
        for (int ni = 0; ni < GRID_; ++ni) {
            float dx = gx - lin28(ni);
            float dx2 = dx * dx;
            for (int nj = 0; nj < GRID_; ++nj, ++n) {
                float dy = gy - lin28(nj);
                float d2 = dx2 + dy * dy;
                bool after = (d2 > pd) || (d2 == pd && n > pn);
                if (after && (d2 < bd || (d2 == bd && n < bn))) { bd = d2; bn = n; }
            }
        }
        idx[m * NBHD_ + s] = bn;
        pd = bd; pn = bn;
    }
}

// ---------------------------------------------------------------------------
// Kernel 2: per-layer neighbor MLP h_l (2->32->32->16, BN+swish each stage).
// deltas are batch-invariant, so BN stats over (B,784,9) == stats over 7056
// points. One workgroup (1024 threads = 32 waves); block-wide stats with
// wave-per-channel shuffle reductions.
// ---------------------------------------------------------------------------
__global__ __launch_bounds__(1024)
void k_compute_h(const int* __restrict__ idx,
                 const float* W0, const float* b0, const float* be0, const float* g0,
                 const float* W1, const float* b1, const float* be1, const float* g1,
                 const float* W2, const float* b2, const float* be2, const float* g2,
                 float* __restrict__ bufA, float* __restrict__ bufB,
                 float* __restrict__ h_out) {
    const int P = M_ * NBHD_;          // 7056
    const int tid = threadIdx.x;
    __shared__ float s_mean[32], s_scale[32], s_shift[32];

    const float* Ws[3]  = {W0, W1, W2};
    const float* bs[3]  = {b0, b1, b2};
    const float* bes[3] = {be0, be1, be2};
    const float* gs[3]  = {g0, g1, g2};
    const int din_[3]  = {2, 32, 32};
    const int dout_[3] = {32, 32, 16};
    float* zout[3] = {bufA, bufB, h_out};
    const float* zin[3] = {nullptr, bufA, bufB};

#pragma unroll
    for (int s = 0; s < 3; ++s) {
        const int din = din_[s], dout = dout_[s];
        const float* W = Ws[s];
        const float* bb = bs[s];
        float* z = zout[s];
        const float* in = zin[s];
        // phase 1: z = in @ W + b
        for (int p = tid; p < P; p += blockDim.x) {
            float rin[32];
            if (s == 0) {
                int m = p / NBHD_;
                int n = idx[p];
                rin[0] = lin28(n / GRID_) - lin28(m / GRID_);
                rin[1] = lin28(n % GRID_) - lin28(m % GRID_);
            } else {
                for (int i = 0; i < din; ++i) rin[i] = in[p * din + i];
            }
            for (int c = 0; c < dout; ++c) {
                float acc = bb[c];
                for (int i = 0; i < din; ++i) acc += rin[i] * W[i * dout + c];
                z[p * dout + c] = acc;
            }
        }
        __threadfence_block();
        __syncthreads();
        // phase 2: per-channel mean/var (wave w owns channel w)
        int wave = tid >> 5, lane = tid & 31;
        if (wave < dout) {
            float sum = 0.0f, ss = 0.0f;
            for (int p = lane; p < P; p += 32) {
                float x = z[p * dout + wave];
                sum += x; ss += x * x;
            }
            for (int off = 16; off; off >>= 1) {
                sum += __shfl_xor(sum, off, 32);
                ss  += __shfl_xor(ss,  off, 32);
            }
            if (lane == 0) {
                float mean = sum / (float)P;
                float var  = ss / (float)P - mean * mean;
                s_mean[wave]  = mean;
                s_scale[wave] = gs[s][wave] * rsqrtf(var + 1e-5f);
                s_shift[wave] = bes[s][wave];
            }
        }
        __syncthreads();
        // phase 3: BN + swish, in place
        for (int p = tid; p < P; p += blockDim.x) {
            for (int c = 0; c < dout; ++c) {
                float x = z[p * dout + c];
                float y = s_scale[c] * (x - s_mean[c]) + s_shift[c];
                z[p * dout + c] = y / (1.0f + __expf(-y));
            }
        }
        __threadfence_block();
        __syncthreads();
    }
}

// ---------------------------------------------------------------------------
// Kernel 3: RBF encode. v0[b,m] = [gx, gy, h0, h1/(h0+1e-8)]
// Transcendental-bound: 64*784*400 exps; 2 MACs per exp -> VALU path.
// ---------------------------------------------------------------------------
__global__ __launch_bounds__(256)
void k_stage1(const float* __restrict__ ctx_coords,  // [B][400][2]
              const float* __restrict__ ctx_values,  // [B][400]
              const float* __restrict__ raw_ls, const float* __restrict__ raw_os,
              float* __restrict__ v0) {              // [B][M_][4]
    __shared__ float scx[NCTX], scy[NCTX], scv[NCTX];
    int b = blockIdx.y;
    for (int n = threadIdx.x; n < NCTX; n += blockDim.x) {
        scx[n] = ctx_coords[((size_t)b * NCTX + n) * 2 + 0];
        scy[n] = ctx_coords[((size_t)b * NCTX + n) * 2 + 1];
        scv[n] = ctx_values[(size_t)b * NCTX + n];
    }
    __syncthreads();
    int m = blockIdx.x * blockDim.x + threadIdx.x;
    if (m >= M_) return;
    float ls  = softplusf(raw_ls[0]);
    float osf = softplusf(raw_os[0]);
    float inv = -0.5f / (ls * ls);
    float gx = lin28(m / GRID_), gy = lin28(m % GRID_);
    float h0 = 0.0f, h1 = 0.0f;
    for (int n = 0; n < NCTX; ++n) {
        float dx = gx - scx[n], dy = gy - scy[n];
        float w = osf * __expf(inv * (dx * dx + dy * dy));
        h0 += w;
        h1 += w * scv[n];
    }
    float* vp = v0 + ((size_t)b * M_ + m) * 4;
    vp[0] = gx; vp[1] = gy; vp[2] = h0; vp[3] = h1 / (h0 + 1e-8f);
}

// ---------------------------------------------------------------------------
// Kernel 4: per-layer pre-contraction  BmatT[m][c][kk] = sum_o h[m,k,o]*W[i*16+o,c]
// (kk = k*cin+i, zero-padded to Kp / coutp). Stored f16, K-contiguous so a
// WMMA B fragment is one aligned 32-byte load.
// ---------------------------------------------------------------------------
__global__ void k_build_bmat(const float* __restrict__ h,    // [M_*9][16]
                             const float* __restrict__ W,    // [16*cin][cout]
                             _Float16* __restrict__ BmatT,   // [M_][coutp][Kp]
                             int cin, int cout, int coutp, int Kp) {
    int m = blockIdx.x;
    int tot = coutp * Kp;
    int K9 = NBHD_ * cin;
    for (int e = threadIdx.x; e < tot; e += blockDim.x) {
        int c = e / Kp, kk = e % Kp;
        float val = 0.0f;
        if (c < cout && kk < K9) {
            int k = kk / cin, i = kk % cin;
            const float* hp = h + ((size_t)m * NBHD_ + k) * 16;
            const float* wp = W + (size_t)(i * 16) * cout + c;
#pragma unroll
            for (int o = 0; o < 16; ++o) val += hp[o] * wp[(size_t)o * cout];
        }
        BmatT[((size_t)m * coutp + c) * Kp + kk] = (_Float16)val;
    }
}

// ---------------------------------------------------------------------------
// Kernel 5: point-conv GEMM via WMMA, compile-time specialized per layer.
// One block per grid point m: 64 batch rows (4 waves x 16) x K=9*CIN (padded)
// x COUT (padded). The per-m B tile (2-10KB, contiguous) is staged into LDS
// with CDNA5 async global->LDS loads (ASYNCcnt) overlapping the f32->f16
// neighbor gather; all 4 waves then read B fragments from LDS (4x reuse).
// Fragment layouts per CDNA5 ISA 7.12.2.
// ---------------------------------------------------------------------------
template<int CIN, int COUT, bool RELU>
__global__ __launch_bounds__(128)
void k_pointconv(const float* __restrict__ v_in,      // [B][M_][CIN]
                 float* __restrict__ v_out,           // [B][M_][COUT]
                 const int* __restrict__ idx,
                 const _Float16* __restrict__ BmatT,  // [M_][COUTP][KP]
                 const float* __restrict__ bias) {
    constexpr int K9      = NBHD_ * CIN;
    constexpr int KP      = (K9 + 31) & ~31;
    constexpr int COUTP   = (COUT + 15) & ~15;
    constexpr int NTILES  = COUTP / 16;
    constexpr int NCHUNKS = KP / 32;
    constexpr int BM_BYTES = COUTP * KP * 2;   // multiple of 16

    __shared__ alignas(32) _Float16 nbv[64 * KP];     // gathered neighbor rows
    __shared__ alignas(32) _Float16 sB[COUTP * KP];   // staged B tile
    __shared__ int idxm[NBHD_];

    const int m = blockIdx.x;
    const int tid = threadIdx.x;

    // --- async global->LDS stage of this grid point's B tile ---------------
    {
        const char* gsrc = (const char*)(BmatT + (size_t)m * (COUTP * KP));
        char* ldst = (char*)sB;
        for (int off = tid * 16; off < BM_BYTES; off += 128 * 16) {
            uint32_t laddr = (uint32_t)(uintptr_t)(ldst + off);
            uint64_t gaddr = (uint64_t)(uintptr_t)(gsrc + off);
            asm volatile("global_load_async_to_lds_b128 %0, %1, off"
                         :: "v"(laddr), "v"(gaddr) : "memory");
        }
    }
    if (tid < NBHD_) idxm[tid] = idx[m * NBHD_ + tid];
    __syncthreads();

    // --- gather neighbor features (f32 -> f16) into LDS (overlaps async) ---
    for (int e = tid; e < 64 * KP; e += 128) {
        int b = e / KP, kk = e % KP;
        _Float16 val = (_Float16)0.0f;
        if (kk < K9) {
            int k = kk / CIN, i = kk % CIN;
            val = (_Float16)v_in[((size_t)b * M_ + idxm[k]) * CIN + i];
        }
        nbv[b * KP + kk] = val;
    }
    asm volatile("s_wait_asynccnt 0x0" ::: "memory");
    __syncthreads();

    // --- WMMA: 64 x KP x COUTP ---------------------------------------------
    const int wave = tid >> 5, lane = tid & 31;
    const int rowb = wave * 16 + (lane & 15);  // A-matrix row = batch index
    const int hi = lane >> 4;

    v8f acc0 = {}; v8f acc1 = {};
#pragma unroll
    for (int ch = 0; ch < NCHUNKS; ++ch) {
        // A fragment: 16x32 f16, lane = row M, K contiguous by lane[4]
        v16h a;
#pragma unroll
        for (int r = 0; r < 8; ++r) {
            int kb = ch * 32 + hi * 16 + (r << 1);
            a[2 * r]     = nbv[rowb * KP + kb];
            a[2 * r + 1] = nbv[rowb * KP + kb + 1];
        }
        // B fragments from LDS: 32x16 f16, lane = col N, K group by lane[4]
        int kbase = ch * 32 + hi * 16;
        {
            int c = (lane & 15);
            v16h bf = *(const v16h*)(sB + c * KP + kbase);
            acc0 = __builtin_amdgcn_wmma_f32_16x16x32_f16(
                false, a, false, bf, (short)0, acc0, false, false);
        }
        if constexpr (NTILES > 1) {
            int c = 16 + (lane & 15);
            v16h bf = *(const v16h*)(sB + c * KP + kbase);
            acc1 = __builtin_amdgcn_wmma_f32_16x16x32_f16(
                false, a, false, bf, (short)0, acc1, false, false);
        }
    }
    // D layout: N = lane&15, M = r + 8*hi
#pragma unroll
    for (int t = 0; t < NTILES; ++t) {
        int c = t * 16 + (lane & 15);
        if (c < COUT) {
#pragma unroll
            for (int r = 0; r < 8; ++r) {
                int bout = wave * 16 + r + 8 * hi;
                float av = (t == 1) ? acc1[r] : acc0[r];
                float val = (av + bias[c]) / 9.0f;
                if (RELU) val = fmaxf(val, 0.0f);
                v_out[((size_t)bout * M_ + m) * COUT + c] = val;
            }
        }
    }
}

// ---------------------------------------------------------------------------
// Kernel 6: f = [f_mu, softplus(f_sigma)]
// ---------------------------------------------------------------------------
__global__ void k_finalize_f(const float* __restrict__ v, float* __restrict__ f) {
    int i = blockIdx.x * blockDim.x + threadIdx.x;
    if (i >= B_ * M_) return;
    f[i * 2 + 0] = v[i * 2 + 0];
    f[i * 2 + 1] = softplusf(v[i * 2 + 1]);
}

// ---------------------------------------------------------------------------
// Kernel 7: RBF decode. mu -> d_out[0:B*784); diag(sigma) entries only
// (rest of the 157MB output zeroed by memset). Transcendental-bound.
// ---------------------------------------------------------------------------
__global__ __launch_bounds__(256)
void k_final(const float* __restrict__ tgt_coords,  // [B][784][2]
             const float* __restrict__ f,           // [B][784][2]
             const float* __restrict__ raw_ls, const float* __restrict__ raw_os,
             float* __restrict__ out) {
    __shared__ float sf0[M_], sf1[M_], slin[GRID_];
    int b = blockIdx.y;
    for (int mm = threadIdx.x; mm < M_; mm += blockDim.x) {
        sf0[mm] = f[((size_t)b * M_ + mm) * 2 + 0];
        sf1[mm] = f[((size_t)b * M_ + mm) * 2 + 1];
    }
    if (threadIdx.x < GRID_) slin[threadIdx.x] = lin28(threadIdx.x);
    __syncthreads();
    int t = blockIdx.x * blockDim.x + threadIdx.x;
    if (t >= NTGT) return;
    float ls  = softplusf(raw_ls[0]);
    float osf = softplusf(raw_os[0]);
    float inv = -0.5f / (ls * ls);
    float tx = tgt_coords[((size_t)b * NTGT + t) * 2 + 0];
    float ty = tgt_coords[((size_t)b * NTGT + t) * 2 + 1];
    float mu = 0.0f, sg = 0.0f;
    int mm = 0;
    for (int mi = 0; mi < GRID_; ++mi) {
        float dx = tx - slin[mi];
        float dx2 = dx * dx;
        for (int mj = 0; mj < GRID_; ++mj, ++mm) {
            float dy = ty - slin[mj];
            float w = osf * __expf(inv * (dx2 + dy * dy));
            mu += w * sf0[mm];
            sg += w * sf1[mm];
        }
    }
    out[(size_t)b * NTGT + t] = mu;
    out[(size_t)B_ * NTGT + (size_t)b * NTGT * NTGT + (size_t)t * NTGT + t] = sg;
}

// ---------------------------------------------------------------------------
// Host orchestration. Input order = jax pytree flatten (sorted dict keys):
//   0 ctx_coords, 1 ctx_values, 2 tgt_coords,
//   per layer l (base 3+14l): W, b, then wn stage s (+2+4s): {W, b, be, g},
//   59/60 psi.raw_ls/raw_os, 61/62 psi_rho.raw_ls/raw_os.
// Workspace use: ~36.8 MB.
// ---------------------------------------------------------------------------
extern "C" void kernel_launch(void* const* d_in, const int* in_sizes, int n_in,
                              void* d_out, int out_size, void* d_ws, size_t ws_size,
                              hipStream_t stream) {
    (void)in_sizes; (void)n_in; (void)ws_size;
    const float* ctx_coords = (const float*)d_in[0];
    const float* ctx_values = (const float*)d_in[1];
    const float* tgt_coords = (const float*)d_in[2];
    auto L = [&](int l, int j) -> const float* { return (const float*)d_in[3 + 14 * l + j]; };
    const float* psi_ls = (const float*)d_in[59];
    const float* psi_os = (const float*)d_in[60];
    const float* rho_ls = (const float*)d_in[61];
    const float* rho_os = (const float*)d_in[62];

    float* ws = (float*)d_ws;
    int*   idx  = (int*)ws;                          // 7056 ints
    float* hbuf = ws + 7056;                         // 4 * 7056*16
    float* bufA = hbuf + 4 * 7056 * 16;              // 4 * 7056*32
    float* bufB = bufA + 4 * 7056 * 32;              // 4 * 7056*32
    float* va   = bufB + 4 * 7056 * 32;              // 64*784*32
    float* vb   = va + 64 * 784 * 32;                // 64*784*32
    float* fbuf = vb + 64 * 784 * 32;                // 64*784*2
    _Float16* Bm = (_Float16*)(fbuf + 64 * 784 * 2); // 784*32*288 f16 (32B-aligned)

    // zero the 157MB output (diag matrices are mostly zeros)
    hipMemsetAsync(d_out, 0, (size_t)out_size * sizeof(float), stream);

    k_geometry<<<8, 128, 0, stream>>>(idx);

    for (int l = 0; l < 4; ++l) {
        k_compute_h<<<1, 1024, 0, stream>>>(idx,
            L(l, 2),  L(l, 3),  L(l, 4),  L(l, 5),
            L(l, 6),  L(l, 7),  L(l, 8),  L(l, 9),
            L(l, 10), L(l, 11), L(l, 12), L(l, 13),
            bufA + l * 7056 * 32, bufB + l * 7056 * 32, hbuf + l * 7056 * 16);
    }

    k_stage1<<<dim3(4, B_), 256, 0, stream>>>(ctx_coords, ctx_values, psi_ls, psi_os, va);

    // layer shapes: (cin,cout) = (4,16) (16,32) (32,16) (16,2)
    k_build_bmat<<<M_, 128, 0, stream>>>(hbuf + 0 * 7056 * 16, L(0, 0), Bm, 4, 16, 16, 64);
    k_pointconv<4, 16, true><<<M_, 128, 0, stream>>>(va, vb, idx, Bm, L(0, 1));

    k_build_bmat<<<M_, 128, 0, stream>>>(hbuf + 1 * 7056 * 16, L(1, 0), Bm, 16, 32, 32, 160);
    k_pointconv<16, 32, true><<<M_, 128, 0, stream>>>(vb, va, idx, Bm, L(1, 1));

    k_build_bmat<<<M_, 128, 0, stream>>>(hbuf + 2 * 7056 * 16, L(2, 0), Bm, 32, 16, 16, 288);
    k_pointconv<32, 16, true><<<M_, 128, 0, stream>>>(va, vb, idx, Bm, L(2, 1));

    k_build_bmat<<<M_, 128, 0, stream>>>(hbuf + 3 * 7056 * 16, L(3, 0), Bm, 16, 2, 16, 160);
    k_pointconv<16, 2, false><<<M_, 128, 0, stream>>>(vb, va, idx, Bm, L(3, 1));

    // final v (stride 2) lives in va
    k_finalize_f<<<(B_ * M_ + 255) / 256, 256, 0, stream>>>(va, fbuf);
    k_final<<<dim3(4, B_), 256, 0, stream>>>(tgt_coords, fbuf, rho_ls, rho_os, (float*)d_out);
}
